// DNNF_21861383537314
// MI455X (gfx1250) — matrix operand order repaired and checked by reading, not compiled, not run on hardware
//
#include <hip/hip_runtime.h>
#include <hip/hip_bf16.h>
#include <math.h>

typedef __attribute__((ext_vector_type(16))) __bf16 v16bf;
typedef __attribute__((ext_vector_type(8)))  __bf16 v8bf;
typedef __attribute__((ext_vector_type(8)))  float  v8f;

#define INPUT_DIM 128
#define BATCH     4096
#define NF        256
#define L_LITS    10752
#define N_CONJ    2688
#define PADCOLS   12288   // padded literal columns: 64*(32+48+48+64)

#define LOG2E     1.4426950408889634f

// ---- fast, branchless transcendental helpers (hardware v_exp/v_rcp/v_sqrt) ----
__device__ __forceinline__ float fast_exp(float x) {
    return __builtin_amdgcn_exp2f(x * LOG2E);
}
// tanh(x) = 1 - 2/(e^{2x}+1); saturates cleanly via rcp (inf -> +1, 0 -> -1), no NaN path.
__device__ __forceinline__ float fast_tanh(float x) {
    float e = __builtin_amdgcn_exp2f(x * (2.0f * LOG2E));
    return 1.0f - 2.0f * __builtin_amdgcn_rcpf(e + 1.0f);
}

// ---- fixed DNNF structure (from _build_structure constants) ----
__device__ __forceinline__ int nc_of(int f) {
    const int t[4] = {6, 9, 12, 15};
    return t[f >> 6];
}
__device__ __forceinline__ int conj_start_of(int f) {
    const int P[4] = {0, 384, 960, 1728};   // 64*{0,6,6+9,6+9+12}
    return P[f >> 6] + (f & 63) * nc_of(f);
}
// padded-column start per formula (literal tiles rounded up to 16)
__device__ __forceinline__ int pstart_of(int f) {
    const int Q[4]  = {0, 2048, 5120, 8192}; // 64*{0,32,32+48,32+48+48}
    const int pt[4] = {32, 48, 48, 64};
    return Q[f >> 6] + (f & 63) * pt[f >> 6];
}
__device__ __forceinline__ int ptiles_of(int f) {
    const int p[4] = {2, 3, 3, 4};
    return p[f >> 6];
}

// ================= prep kernels =================

// x -> bf16, x^2 -> bf16
__global__ __launch_bounds__(256) void prep_x(const float* __restrict__ x,
                                              __bf16* __restrict__ xbf,
                                              __bf16* __restrict__ xsq, int n) {
    int i = blockIdx.x * 256 + threadIdx.x;
    if (i < n) {
        float v = x[i];
        xbf[i] = (__bf16)v;
        xsq[i] = (__bf16)(v * v);
    }
}

// masked weight -> padded, literal-major bf16: wbf[pc*128 + k]; padding columns are zero.
__global__ __launch_bounds__(256) void prep_w(const float* __restrict__ W,
                                              const float* __restrict__ lm,
                                              const int* __restrict__ f_of_l,
                                              __bf16* __restrict__ wbf) {
    int i = blockIdx.x * 256 + threadIdx.x;       // i in [0, PADCOLS*128)
    if (i >= PADCOLS * INPUT_DIM) return;
    int pc = i >> 7, k = i & 127;
    const int Q[4]   = {0, 2048, 5120, 8192};
    const int pt[4]  = {32, 48, 48, 64};
    const int ncs[4] = {6, 9, 12, 15};
    int g = (pc >= 8192) ? 3 : (pc >= 5120) ? 2 : (pc >= 2048) ? 1 : 0;
    int rel = pc - Q[g];
    int within = rel / pt[g];
    int localcol = rel - within * pt[g];
    int f = (g << 6) + within;
    float val = 0.0f;
    if (localcol < 4 * ncs[g]) {
        int lit = 4 * conj_start_of(f) + localcol;
        int ff = f_of_l[lit];                     // == f; use the actual input mapping
        float m = (fabsf(lm[k * NF + ff]) > 1.0f) ? 1.0f : 0.0f;
        val = W[(size_t)k * L_LITS + lit] * m;
    }
    wbf[(size_t)pc * INPUT_DIM + k] = (__bf16)val;
}

// RBF B matrix (K=256: first 128 = -2*s2*mu, last 128 = s2) + constant c2[f] = sum s2*mu^2
__global__ __launch_bounds__(128) void prep_rbf(const float* __restrict__ mu,
                                                const float* __restrict__ sigma,
                                                __bf16* __restrict__ rbfB,
                                                float* __restrict__ c2) {
    __shared__ float sred[128];
    int f = blockIdx.x, k = threadIdx.x;
    float s = sigma[f * INPUT_DIM + k];
    float s2 = s * s;
    float m = mu[f * INPUT_DIM + k];
    rbfB[(size_t)f * 256 + k]       = (__bf16)(-2.0f * s2 * m);
    rbfB[(size_t)f * 256 + 128 + k] = (__bf16)(s2);
    sred[k] = s2 * m * m;
    __syncthreads();
    for (int sr = 64; sr > 0; sr >>= 1) {
        if (k < sr) sred[k] += sred[k + sr];
        __syncthreads();
    }
    if (k == 0) c2[f] = sred[0];
}

// ================= main DNNF kernel =================
// grid (BATCH/32, 2): 32-row batch tile per block (2 accumulators share each B frag),
// formulas parity-split over blockIdx.y; 8 waves/block, each wave does 16 formulas.
__global__ __launch_bounds__(256) void dnnf_main(const __bf16* __restrict__ xbf,
                                                 const __bf16* __restrict__ wbf,
                                                 const float* __restrict__ bias,
                                                 const int* __restrict__ c_of_l,
                                                 float* __restrict__ dnnf) {
    __shared__ float csum[8][32][16];
    const int wave = threadIdx.x >> 5;
    const int lane = threadIdx.x & 31;
    const int half = lane >> 4;          // 0: lanes 0-15, 1: lanes 16-31
    const int m0 = blockIdx.x * 32;

    // Preload A fragments for both 16-row tiles, K = 0..127 in 4 chunks of 32.
    // 16-bit A layout: lane<16 holds K {0..7, 16..23}; lane>=16 holds K {8..15, 24..31}.
    v16bf afrag[2][4];
#pragma unroll
    for (int mt = 0; mt < 2; ++mt) {
        const __bf16* xrow = xbf + (size_t)(m0 + 16 * mt + (lane & 15)) * INPUT_DIM;
#pragma unroll
        for (int kk = 0; kk < 4; ++kk) {
            int k0 = kk * 32;
            v8bf lo = *(const v8bf*)(xrow + k0 + half * 8);
            v8bf hi = *(const v8bf*)(xrow + k0 + 16 + half * 8);
#pragma unroll
            for (int i = 0; i < 8; ++i) { afrag[mt][kk][i] = lo[i]; afrag[mt][kk][i + 8] = hi[i]; }
        }
    }

    float (*my)[16] = csum[wave];
    float* myf = &csum[wave][0][0];

    for (int it = 0; it < 16; ++it) {
        const int f = ((wave + 8 * it) << 1) + blockIdx.y;   // parity-interleaved formulas
        const int nc = nc_of(f);
        const int cstart = conj_start_of(f);
        const int lstart = 4 * cstart;
        const int pbase  = pstart_of(f);
        const int ptile  = ptiles_of(f);

        // zero per-wave conj accumulator (32 rows x 16 conj)
        for (int i = lane; i < 512; i += 32) myf[i] = 0.0f;
        __builtin_amdgcn_wave_barrier();

        for (int t = 0; t < ptile; ++t) {
            const int lc = t * 16 + (lane & 15);             // local column within formula
            const __bf16* wcol = wbf + (size_t)(pbase + lc) * INPUT_DIM;
            // Unconditional padded B loads: 8x global_load_b128, clause-able.
            v8bf blo[4], bhi[4];
#pragma unroll
            for (int kk = 0; kk < 4; ++kk) {
                blo[kk] = *(const v8bf*)(wcol + kk * 32 + half * 8);
                bhi[kk] = *(const v8bf*)(wcol + kk * 32 + 16 + half * 8);
            }
            v8f acc0 = {}, acc1 = {};
#pragma unroll
            for (int kk = 0; kk < 4; ++kk) {
                v16bf bfrag;
#pragma unroll
                for (int i = 0; i < 8; ++i) { bfrag[i] = blo[kk][i]; bfrag[i + 8] = bhi[kk][i]; }
                acc0 = __builtin_amdgcn_wmma_f32_16x16x32_bf16(
                    false, afrag[0][kk], false, bfrag, (short)0, acc0, false, false);
                acc1 = __builtin_amdgcn_wmma_f32_16x16x32_bf16(
                    false, afrag[1][kk], false, bfrag, (short)0, acc1, false, false);
            }
            if (lc < 4 * nc) {                               // scatter only real literals
                const int lit = lstart + lc;
                const float bv = bias[lit];
                const int cl = c_of_l[lit] - cstart;         // local conj index (< nc <= 15)
#pragma unroll
                for (int r = 0; r < 8; ++r) {
                    const int m = r + half * 8;              // C layout: lanes>=16 hold M=r+8
                    atomicAdd(&my[m][cl],      fast_tanh(acc0[r] + bv));
                    atomicAdd(&my[m + 16][cl], fast_tanh(acc1[r] + bv));
                }
            }
        }
        asm volatile("s_wait_dscnt 0" ::: "memory");         // same-wave LDS atomics visible
        __builtin_amdgcn_wave_barrier();

        // conj -> formula: out_conj = tanh(sum - depth + 1.5); reduce; tanh(+nc-1.5)
        const int j = lane & 15;
        const int t3 = nc / 3;
        const float depth = (j < t3) ? 2.0f : ((j < 2 * t3) ? 4.0f : 6.0f);
#pragma unroll
        for (int r = 0; r < 8; ++r) {
#pragma unroll
            for (int mt = 0; mt < 2; ++mt) {
                const int m = r + half * 8 + 16 * mt;
                float v = (j < nc) ? fast_tanh(my[m][j] - depth + 1.5f) : 0.0f;
#pragma unroll
                for (int off = 1; off < 16; off <<= 1) v += __shfl_xor(v, off, 32);
                if (j == 0) dnnf[(size_t)(m0 + m) * NF + f] = fast_tanh(v + (float)nc - 1.5f);
            }
        }
        __builtin_amdgcn_wave_barrier();
    }
}

// ================= RBF kernel =================
// norm^2 = [x, x^2] . [-2*s2*mu ; s2] + c2[f];  loc_raw = exp(-sqrt(max(norm2,0)))
// grid (BATCH/16, 2), 8 waves/block; wave handles formula tile f0 = (blockIdx.y*8+wave)*16.
__global__ __launch_bounds__(256) void rbf_kernel(const __bf16* __restrict__ xbf,
                                                  const __bf16* __restrict__ xsq,
                                                  const __bf16* __restrict__ rbfB,
                                                  const float* __restrict__ c2,
                                                  float* __restrict__ loc) {
    const int wave = threadIdx.x >> 5;
    const int lane = threadIdx.x & 31;
    const int half = lane >> 4;
    const int m0 = blockIdx.x * 16;
    const int f0 = (blockIdx.y * 8 + wave) * 16;
    const int row = m0 + (lane & 15);
    const int fcol = f0 + (lane & 15);

    const __bf16* xrow = xbf + (size_t)row * INPUT_DIM;
    const __bf16* qrow = xsq + (size_t)row * INPUT_DIM;
    const __bf16* bcol = rbfB + (size_t)fcol * 256;

    v8f acc = {};
#pragma unroll
    for (int kk = 0; kk < 8; ++kk) {                 // K = 256 in chunks of 32
        const int k0 = kk * 32;
        v16bf af, bf_;
        const int kA0 = k0 + half * 8;
        const int kA1 = k0 + 16 + half * 8;
        v8bf a0 = (kA0 < 128) ? *(const v8bf*)(xrow + kA0) : *(const v8bf*)(qrow + kA0 - 128);
        v8bf a1 = (kA1 < 128) ? *(const v8bf*)(xrow + kA1) : *(const v8bf*)(qrow + kA1 - 128);
        v8bf b0 = *(const v8bf*)(bcol + k0 + half * 8);
        v8bf b1 = *(const v8bf*)(bcol + k0 + 16 + half * 8);
#pragma unroll
        for (int i = 0; i < 8; ++i) {
            af[i] = a0[i]; af[i + 8] = a1[i];
            bf_[i] = b0[i]; bf_[i + 8] = b1[i];
        }
        acc = __builtin_amdgcn_wmma_f32_16x16x32_bf16(
            false, af, false, bf_, (short)0, acc, false, false);
    }
    const float cc = c2[fcol];
#pragma unroll
    for (int r = 0; r < 8; ++r) {
        const int m = r + half * 8;
        float n2 = fmaxf(acc[r] + cc, 0.0f);
        loc[(size_t)(m0 + m) * NF + fcol] = fast_exp(-__builtin_amdgcn_sqrtf(n2));
    }
}

// ================= softmax over formulas + combine =================
__global__ __launch_bounds__(256) void softmax_combine(const float* __restrict__ loc,
                                                       const float* __restrict__ dnnf,
                                                       const float* __restrict__ temperature,
                                                       float* __restrict__ out) {
    __shared__ float red[256];
    const int b = blockIdx.x, f = threadIdx.x;
    const float T = temperature[0];
    const float sig = __builtin_amdgcn_rcpf(1.0f + fast_exp(-T));
    const float v = sig * loc[(size_t)b * NF + f];
    red[f] = v;
    __syncthreads();
    for (int s = 128; s > 0; s >>= 1) {
        if (f < s) red[f] = fmaxf(red[f], red[f + s]);
        __syncthreads();
    }
    const float mx = red[0];
    __syncthreads();
    const float e = fast_exp(v - mx);
    red[f] = e;
    __syncthreads();
    for (int s = 128; s > 0; s >>= 1) {
        if (f < s) red[f] += red[f + s];
        __syncthreads();
    }
    out[(size_t)b * NF + f] = dnnf[(size_t)b * NF + f] * (e * __builtin_amdgcn_rcpf(red[0]));
}

// ================= launcher =================
extern "C" void kernel_launch(void* const* d_in, const int* in_sizes, int n_in,
                              void* d_out, int out_size, void* d_ws, size_t ws_size,
                              hipStream_t stream) {
    const float* x    = (const float*)d_in[0];
    const float* W    = (const float*)d_in[1];
    const float* bias = (const float*)d_in[2];
    const float* lm   = (const float*)d_in[3];
    const float* mu   = (const float*)d_in[4];
    const float* sig  = (const float*)d_in[5];
    const float* temp = (const float*)d_in[6];
    const int* f_of_l = (const int*)d_in[7];
    const int* c_of_l = (const int*)d_in[8];
    float* out = (float*)d_out;

    // workspace carve-up (all offsets multiple of 64 B)
    char* ws = (char*)d_ws;
    size_t off = 0;
    __bf16* xbf  = (__bf16*)(ws + off); off += (size_t)BATCH * INPUT_DIM * 2;      // 1 MB
    __bf16* xsq  = (__bf16*)(ws + off); off += (size_t)BATCH * INPUT_DIM * 2;      // 1 MB
    __bf16* wbf  = (__bf16*)(ws + off); off += (size_t)PADCOLS * INPUT_DIM * 2;    // 3 MB
    __bf16* rbfB = (__bf16*)(ws + off); off += (size_t)NF * 256 * 2;               // 128 KB
    float*  c2   = (float*)(ws + off);  off += (size_t)NF * 4 + 192;               // 1 KB
    float*  dnnf = (float*)(ws + off);  off += (size_t)BATCH * NF * 4;             // 4 MB
    float*  loc  = (float*)(ws + off);  off += (size_t)BATCH * NF * 4;             // 4 MB

    // prep
    {
        int n = BATCH * INPUT_DIM;
        prep_x<<<(n + 255) / 256, 256, 0, stream>>>(x, xbf, xsq, n);
    }
    {
        int n = PADCOLS * INPUT_DIM;
        prep_w<<<(n + 255) / 256, 256, 0, stream>>>(W, lm, f_of_l, wbf);
    }
    prep_rbf<<<NF, 128, 0, stream>>>(mu, sig, rbfB, c2);

    // main DNNF GEMM + fused segment reductions (32-row tiles, formula parity split)
    dnnf_main<<<dim3(BATCH / 32, 2), 256, 0, stream>>>(xbf, wbf, bias, c_of_l, dnnf);

    // RBF localization GEMM
    rbf_kernel<<<dim3(BATCH / 16, 2), 256, 0, stream>>>(xbf, xsq, rbfB, c2, loc);

    // softmax over formulas + final product
    softmax_combine<<<BATCH, 256, 0, stream>>>(loc, dnnf, temp, out);
}
